// SimpleAttention_30648886624923
// MI455X (gfx1250) — matrix operand order repaired and checked by reading, not compiled, not run on hardware
//
#include <hip/hip_runtime.h>

typedef __attribute__((ext_vector_type(16))) _Float16 v16h;
typedef __attribute__((ext_vector_type(2)))  _Float16 v2h;
typedef __attribute__((ext_vector_type(8)))  float    v8f;
typedef __attribute__((ext_vector_type(4)))  int      v4i;

namespace {
constexpr int   kB   = 8;
constexpr int   kS   = 2048;
constexpr int   kD   = 512;
constexpr int   BQ   = 128;   // query rows per block (8 waves x 16)
constexpr int   BK   = 32;    // key tile size
constexpr int   SXS  = 520;   // sX row stride (halfs): 1040B = 65*16B, banks ln*4
constexpr int   SVS  = 40;    // sVt row stride (halfs): 80B = 5*16B, banks ln*20
constexpr int   SPS  = 36;    // sP row stride (halfs): 72B, banks ln*18
constexpr float kNeg = -1e9f;
constexpr float kNegInf = -3.0e38f;
}

#if __has_builtin(__builtin_amdgcn_global_load_async_to_lds_b128)
#define HAVE_ASYNC_LDS 1
#else
#define HAVE_ASYNC_LDS 0
#endif

static __device__ __forceinline__ v8f wmma_f16(v16h a, v16h b, v8f c) {
  // D = A(16x32 f16) * B(32x16 f16) + C(16x16 f32)
  return __builtin_amdgcn_wmma_f32_16x16x32_f16(false, a, false, b,
                                                (short)0, c, false, false);
}

// ---------------------------------------------------------------------------
// Shared per-key-tile flash-attention compute (identical for both kernels).
// sXb : K tile [key][d] (padded SXS), sVtb : V tile [d][key] (padded SVS),
// sPw : this wave's P scratch.
// ---------------------------------------------------------------------------
static __device__ __forceinline__ void tile_compute(
    const _Float16* __restrict__ sXb, const _Float16* __restrict__ sVtb,
    _Float16* __restrict__ sPw, const int* __restrict__ maskRow,
    const v16h* __restrict__ q, v8f* __restrict__ o,
    float* __restrict__ mr, float* __restrict__ lr,
    int k0, int qrow0, int ln, int hi) {
  // ---- S = Q K^T : two 16x16 accumulators over 16 K-chunks of 32 ----
  v8f sc[2];
  #pragma unroll
  for (int nt = 0; nt < 2; ++nt) {
    #pragma unroll
    for (int e = 0; e < 8; ++e) sc[nt][e] = 0.f;
    const int key = nt * 16 + ln;       // B-layout: N = ln
    #pragma unroll
    for (int c = 0; c < 16; ++c) {
      v16h bk;                          // B: K-dim pair = table order (matches A)
      #pragma unroll
      for (int v = 0; v < 8; ++v) {
        const int d = c * 32 + (v < 4 ? v * 2 : 16 + (v - 4) * 2) + hi * 8;
        bk[2 * v]     = sXb[key * SXS + d];
        bk[2 * v + 1] = sXb[key * SXS + d + 1];
      }
      sc[nt] = wmma_f16(q[c], bk, sc[nt]);
    }
  }

  // ---- masking (additive, like reference) + row max ----
  const bool needCausal = (k0 + BK - 1) > qrow0;   // wave-uniform
  float mx[8];
  #pragma unroll
  for (int r = 0; r < 8; ++r) mx[r] = kNegInf;
  #pragma unroll
  for (int nt = 0; nt < 2; ++nt) {
    const int key = k0 + nt * 16 + ln;
    const float padAdd = maskRow[key] ? 0.f : kNeg;
    if (needCausal) {
      #pragma unroll
      for (int r = 0; r < 8; ++r) {
        const int qrow = qrow0 + r + hi * 8;
        float s = sc[nt][r] + padAdd + (key > qrow ? kNeg : 0.f);
        sc[nt][r] = s;
        mx[r] = fmaxf(mx[r], s);
      }
    } else {
      #pragma unroll
      for (int r = 0; r < 8; ++r) {
        float s = sc[nt][r] + padAdd;
        sc[nt][r] = s;
        mx[r] = fmaxf(mx[r], s);
      }
    }
  }
  // reduce across the 16 lanes of each half-wave (rows live per-VGPR)
  #pragma unroll
  for (int r = 0; r < 8; ++r) {
    #pragma unroll
    for (int off = 1; off < 16; off <<= 1)
      mx[r] = fmaxf(mx[r], __shfl_xor(mx[r], off));
  }

  float alpha[8];
  #pragma unroll
  for (int r = 0; r < 8; ++r) {
    const float mn = fmaxf(mr[r], mx[r]);
    alpha[r] = __expf(mr[r] - mn);
    mr[r] = mn;
  }

  // ---- P = exp(S - m), row sums, stash P to LDS for layout transpose ----
  float ps[8];
  #pragma unroll
  for (int r = 0; r < 8; ++r) ps[r] = 0.f;
  #pragma unroll
  for (int nt = 0; nt < 2; ++nt) {
    #pragma unroll
    for (int r = 0; r < 8; ++r) {
      const float p = __expf(sc[nt][r] - mr[r]);
      ps[r] += p;
      sPw[(r + hi * 8) * SPS + nt * 16 + ln] = (_Float16)p;
    }
  }
  #pragma unroll
  for (int r = 0; r < 8; ++r) {
    #pragma unroll
    for (int off = 1; off < 16; off <<= 1)
      ps[r] += __shfl_xor(ps[r], off);
    lr[r] = lr[r] * alpha[r] + ps[r];
  }

  // rescale running O by alpha (per-VGPR == per-row)
  #pragma unroll
  for (int t = 0; t < 32; ++t) {
    #pragma unroll
    for (int r = 0; r < 8; ++r) o[t][r] *= alpha[r];
  }

  // ---- read P back in A-layout (C-layout -> A-layout via LDS) ----
  v16h pa;
  #pragma unroll
  for (int v = 0; v < 8; ++v) {
    const int k = (v < 4 ? v * 2 : 16 + (v - 4) * 2) + hi * 8;
    pa[2 * v]     = sPw[ln * SPS + k];
    pa[2 * v + 1] = sPw[ln * SPS + k + 1];
  }

  // ---- O += P * V over 32 D-tiles (contiguous keys in sVt -> b128) ----
  #pragma unroll
  for (int t = 0; t < 32; ++t) {
    v16h bv;
    const int d = t * 16 + ln;
    #pragma unroll
    for (int v = 0; v < 8; ++v) {
      const int key = (v < 4 ? v * 2 : 16 + (v - 4) * 2) + hi * 8;
      bv[2 * v]     = sVtb[d * SVS + key];
      bv[2 * v + 1] = sVtb[d * SVS + key + 1];
    }
    o[t] = wmma_f16(pa, bv, o[t]);
  }
}

static __device__ __forceinline__ void store_output(
    float* __restrict__ Out, const v8f* __restrict__ o,
    const float* __restrict__ lr, int b, int qrow0, int ln, int hi) {
  float inv[8];
  #pragma unroll
  for (int r = 0; r < 8; ++r) inv[r] = 1.f / lr[r];
  #pragma unroll
  for (int t = 0; t < 32; ++t) {
    #pragma unroll
    for (int r = 0; r < 8; ++r) {
      const int qrow = qrow0 + r + hi * 8;
      Out[((size_t)b * kS + qrow) * kD + t * 16 + ln] = o[t][r] * inv[r];
    }
  }
}

// ===========================================================================
// Fallback path (proven): in-kernel f32->f16 conversion, synchronous staging.
// ===========================================================================
__global__ __launch_bounds__(256, 1)
void SimpleAttention_fa_sync(const float* __restrict__ X,
                             const int*   __restrict__ mask,
                             float* __restrict__ Out) {
  __shared__ _Float16 sX [BK * SXS];
  __shared__ _Float16 sVt[kD * SVS];
  __shared__ _Float16 sP [8][16 * SPS];

  const int b    = blockIdx.y;
  const int qb   = blockIdx.x * BQ;
  const int tid  = threadIdx.x;
  const int wave = tid >> 5;
  const int lane = tid & 31;
  const int ln   = lane & 15;
  const int hi   = lane >> 4;

  const float* Xb = X + (size_t)b * kS * kD;
  const int qrow0 = qb + wave * 16;

  v16h q[16];
  {
    const float* qp = Xb + (size_t)(qrow0 + ln) * kD;
    #pragma unroll
    for (int c = 0; c < 16; ++c) {
      #pragma unroll
      for (int v = 0; v < 8; ++v) {
        const int k = (v < 4 ? v * 2 : 16 + (v - 4) * 2) + hi * 8;
        const float2 f = *(const float2*)(qp + c * 32 + k);
        q[c][2 * v]     = (_Float16)f.x;
        q[c][2 * v + 1] = (_Float16)f.y;
      }
    }
  }

  v8f o[32];
  #pragma unroll
  for (int t = 0; t < 32; ++t) {
    #pragma unroll
    for (int e = 0; e < 8; ++e) o[t][e] = 0.f;
  }
  float mr[8], lr[8];
  #pragma unroll
  for (int r = 0; r < 8; ++r) { mr[r] = kNegInf; lr[r] = 0.f; }

  const int nkt = (qb + BQ) / BK;
  for (int kt = 0; kt < nkt; ++kt) {
    const int k0 = kt * BK;
    __syncthreads();
    {
      const float* src = Xb + (size_t)k0 * kD;
      const bool pf = (kt + 1) < nkt;
      for (int i = tid * 4; i < BK * kD; i += 256 * 4) {
        if (pf) __builtin_prefetch(src + BK * kD + i, 0, 1);
        const float4 f = *(const float4*)(src + i);
        const int key = i >> 9;
        const int d   = i & 511;
        sX[key * SXS + d + 0] = (_Float16)f.x;
        sX[key * SXS + d + 1] = (_Float16)f.y;
        sX[key * SXS + d + 2] = (_Float16)f.z;
        sX[key * SXS + d + 3] = (_Float16)f.w;
        sVt[(d + 0) * SVS + key] = (_Float16)f.x;
        sVt[(d + 1) * SVS + key] = (_Float16)f.y;
        sVt[(d + 2) * SVS + key] = (_Float16)f.z;
        sVt[(d + 3) * SVS + key] = (_Float16)f.w;
      }
    }
    __syncthreads();
    if (k0 <= qrow0 + 15)
      tile_compute(sX, sVt, sP[wave], mask + b * kS, q, o, mr, lr,
                   k0, qrow0, ln, hi);
  }
  store_output(Out, o, lr, b, qrow0, ln, hi);
}

// ===========================================================================
// Async path: prep kernel writes f16 X and f16 X^T into workspace; the main
// kernel double-buffers LDS tiles with GLOBAL_LOAD_ASYNC_TO_LDS_B128.
// ===========================================================================
__global__ __launch_bounds__(256)
void SimpleAttention_cvt(const float* __restrict__ X,
                         _Float16* __restrict__ Xh,
                         _Float16* __restrict__ XT) {
  const size_t n = (size_t)kB * kS * kD;
  for (size_t i = (size_t)blockIdx.x * blockDim.x + threadIdx.x; i < n;
       i += (size_t)gridDim.x * blockDim.x) {
    const _Float16 h = (_Float16)X[i];
    Xh[i] = h;
    const int    d = (int)(i % kD);
    const size_t r = i / kD;
    const int    s = (int)(r % kS);
    const int    b = (int)(r / kS);
    XT[((size_t)b * kD + d) * kS + s] = h;
  }
}

#if HAVE_ASYNC_LDS
// Builtin signature (from clang diagnostic): param 0 is a *global* (AS1,
// printed as __device__) pointer to a 128-bit int4 payload, non-const.
typedef __attribute__((address_space(1))) v4i gv4i;
typedef __attribute__((address_space(3))) v4i lv4i;

static __device__ __forceinline__ void async_cp16(const _Float16* g,
                                                  _Float16* l) {
  __builtin_amdgcn_global_load_async_to_lds_b128((gv4i*)g, (lv4i*)l, 0, 0);
}

static __device__ __forceinline__ void wait_async0() {
#if __has_builtin(__builtin_amdgcn_s_wait_asynccnt)
  __builtin_amdgcn_s_wait_asynccnt(0);
#else
  asm volatile("s_wait_asynccnt 0" ::: "memory");
#endif
}

__global__ __launch_bounds__(256, 1)
void SimpleAttention_fa_async(const _Float16* __restrict__ Xh,
                              const _Float16* __restrict__ XT,
                              const int*      __restrict__ mask,
                              float* __restrict__ Out) {
  __shared__ _Float16 sX [2][BK * SXS];   // double-buffered K tile
  __shared__ _Float16 sVt[2][kD * SVS];   // double-buffered V^T tile
  __shared__ _Float16 sP [8][16 * SPS];

  const int b    = blockIdx.y;
  const int qb   = blockIdx.x * BQ;
  const int tid  = threadIdx.x;
  const int wave = tid >> 5;
  const int lane = tid & 31;
  const int ln   = lane & 15;
  const int hi   = lane >> 4;

  const _Float16* Xhb = Xh + (size_t)b * kS * kD;   // [s][d] f16
  const _Float16* XTb = XT + (size_t)b * kD * kS;   // [d][s] f16
  const int qrow0 = qb + wave * 16;

  // Issue the per-thread async copies for one key tile into buffer `bi`.
  auto issue_tile = [&](int k0, int bi) {
    // sX: 32 rows x 512 halfs = 2048 x 16B segments
    #pragma unroll
    for (int t = 0; t < 8; ++t) {
      const int j    = tid + t * 256;
      const int key  = j >> 6;
      const int dseg = (j & 63) * 8;
      async_cp16(Xhb + (size_t)(k0 + key) * kD + dseg,
                 &sX[bi][key * SXS + dseg]);
    }
    // sVt: 512 rows x 32 halfs = 512 x 4 x 16B segments
    #pragma unroll
    for (int t = 0; t < 8; ++t) {
      const int j    = tid + t * 256;
      const int d    = j >> 2;
      const int kseg = (j & 3) * 8;
      async_cp16(XTb + (size_t)d * kS + k0 + kseg,
                 &sVt[bi][d * SVS + kseg]);
    }
  };

  // ---- Q tile -> f16 registers in A-layout (b32 loads from f16 copy) ----
  v16h q[16];
  {
    const _Float16* qp = Xhb + (size_t)(qrow0 + ln) * kD;
    #pragma unroll
    for (int c = 0; c < 16; ++c) {
      #pragma unroll
      for (int v = 0; v < 8; ++v) {
        const int k = (v < 4 ? v * 2 : 16 + (v - 4) * 2) + hi * 8;
        const v2h f = *(const v2h*)(qp + c * 32 + k);
        q[c][2 * v]     = f[0];
        q[c][2 * v + 1] = f[1];
      }
    }
  }

  v8f o[32];
  #pragma unroll
  for (int t = 0; t < 32; ++t) {
    #pragma unroll
    for (int e = 0; e < 8; ++e) o[t][e] = 0.f;
  }
  float mr[8], lr[8];
  #pragma unroll
  for (int r = 0; r < 8; ++r) { mr[r] = kNegInf; lr[r] = 0.f; }

  const int nkt = (qb + BQ) / BK;
  issue_tile(0, 0);                      // prologue: fill buffer 0
  for (int kt = 0; kt < nkt; ++kt) {
    const int k0 = kt * BK;
    const int bi = kt & 1;
    wait_async0();                       // my copies into buf bi complete
    __syncthreads();                     // everyone's copies complete;
                                         // prev iteration's reads done
    if (kt + 1 < nkt) issue_tile(k0 + BK, bi ^ 1);  // overlap with compute
    if (k0 <= qrow0 + 15)
      tile_compute(sX[bi], sVt[bi], sP[wave], mask + b * kS, q, o, mr, lr,
                   k0, qrow0, ln, hi);
  }
  store_output(Out, o, lr, b, qrow0, ln, hi);
}
#endif  // HAVE_ASYNC_LDS

extern "C" void kernel_launch(void* const* d_in, const int* in_sizes, int n_in,
                              void* d_out, int out_size, void* d_ws, size_t ws_size,
                              hipStream_t stream) {
  (void)in_sizes; (void)n_in; (void)out_size;
  const float* X    = (const float*)d_in[0];
  const int*   mask = (const int*)d_in[1];
  float*       Out  = (float*)d_out;
  dim3 grid(kS / BQ, kB);

#if HAVE_ASYNC_LDS
  const size_t halfElems = (size_t)kB * kS * kD;
  if (ws_size >= 2 * halfElems * sizeof(_Float16)) {
    _Float16* Xh = (_Float16*)d_ws;
    _Float16* XT = Xh + halfElems;
    SimpleAttention_cvt<<<dim3(2048), dim3(256), 0, stream>>>(X, Xh, XT);
    SimpleAttention_fa_async<<<grid, dim3(256), 0, stream>>>(Xh, XT, mask, Out);
    return;
  }
#endif
  (void)d_ws; (void)ws_size;
  SimpleAttention_fa_sync<<<grid, dim3(256), 0, stream>>>(X, mask, Out);
}
